// GraphNets_57354993270911
// MI455X (gfx1250) — compile-verified, matrix operand order checked by reading, not run on hardware
//
#include <hip/hip_runtime.h>
#include <hip/hip_bf16.h>

typedef __attribute__((ext_vector_type(16))) _Float16 v16h;
typedef __attribute__((ext_vector_type(8)))  float    v8f;

// ---------------- problem constants (match reference) ----------------
static const int Nn    = 50000;
static const int Ee    = 800000;
static const int Dd    = 64;
static const int DEe   = 32;
static const int DGg   = 64;
static const int Hh    = 80;     // MLP hidden & HEADS*CH
static const int HEADS = 5;
static const int Bb    = 256;
static const int DEPTH = 3;
static const int E2    = Ee + Nn; // edges + self loops
#define EPSBN 1e-5f

// ---------------- small device helpers ----------------
__device__ __forceinline__ float seluf(float x) {
  const float a = 1.6732632423543772f, sc = 1.0507009873554805f;
  return sc * (x > 0.f ? x : a * (expf(x) - 1.f));
}
__device__ __forceinline__ void atomicMaxF(float* addr, float v) {
  int* ai = (int*)addr;
  int old = __float_as_int(*addr);
  while (__int_as_float(old) < v) {
    int assumed = old;
    old = atomicCAS(ai, assumed, __float_as_int(v));
    if (old == assumed) break;
  }
}
__device__ __forceinline__ void atomicMinF(float* addr, float v) {
  int* ai = (int*)addr;
  int old = __float_as_int(*addr);
  while (__int_as_float(old) > v) {
    int assumed = old;
    old = atomicCAS(ai, assumed, __float_as_int(v));
    if (old == assumed) break;
  }
}

// ---------------- weight packing into WMMA B-fragment order ----------------
// Layout: Wp[((kt*NT + nt)*32 + lane)*16 + e]  (16 halfs per lane = one v16h)
__global__ void pack_w_kernel(const float* __restrict__ W, _Float16* __restrict__ Wp,
                              int K, int N, int NT, long total) {
  long id = (long)blockIdx.x * blockDim.x + threadIdx.x;
  if (id >= total) return;
  const int  e    = (int)(id & 15);
  const int  lane = (int)((id >> 4) & 31);
  const long tile = id >> 9;
  const int  nt   = (int)(tile % NT);
  const int  kt   = (int)(tile / NT);
  const int khalf = (lane >> 4) & 1;
  const int ncol  = lane & 15;
  const int vg    = e >> 1;
  const int k = kt * 32 + ((vg >> 2) << 4) + (khalf << 3) + ((vg & 3) << 1) + (e & 1);
  const int n = nt * 16 + ncol;
  Wp[id] = (k < K && n < N) ? (_Float16)W[(size_t)k * N + n] : (_Float16)0.0f;
}

// ---------------- generic WMMA GEMM: C = act(A[MxK] * W[KxN] + b) ----------------
// NT_C compile-time column tiles (N = 16*NT_C); TAIL: K % 32 == 16.
// act: 0 = none, 1 = SELU.  A rows 64B-aligned -> float4 loads legal.
__device__ __forceinline__ void cvt_a_frag(v16h& a, float4 f10, float4 f11,
                                           float4 f20, float4 f21) {
  a[0]  = (_Float16)f10.x; a[1]  = (_Float16)f10.y; a[2]  = (_Float16)f10.z; a[3]  = (_Float16)f10.w;
  a[4]  = (_Float16)f11.x; a[5]  = (_Float16)f11.y; a[6]  = (_Float16)f11.z; a[7]  = (_Float16)f11.w;
  a[8]  = (_Float16)f20.x; a[9]  = (_Float16)f20.y; a[10] = (_Float16)f20.z; a[11] = (_Float16)f20.w;
  a[12] = (_Float16)f21.x; a[13] = (_Float16)f21.y; a[14] = (_Float16)f21.z; a[15] = (_Float16)f21.w;
}

template <int NT_C, bool TAIL>
__global__ __launch_bounds__(256) void gemm_wmma_kernel(
    const float* __restrict__ A, const _Float16* __restrict__ Wp,
    const float* __restrict__ bias, float* __restrict__ C,
    int M, int K, int act)
{
  const int N     = NT_C * 16;
  const int lane  = threadIdx.x & 31;
  const int wave  = threadIdx.x >> 5;
  const int row0  = blockIdx.x * 128 + wave * 16;
  const int mrow  = row0 + (lane & 15);
  const int aRow  = (mrow < M) ? mrow : (M - 1);
  const int khalf = (lane >> 4) & 1;
  const int ncol  = lane & 15;

  const float* __restrict__ Arow = A + (size_t)aRow * K + khalf * 8;

  v8f zero = {};
  v8f acc[NT_C];
#pragma unroll
  for (int t = 0; t < NT_C; ++t) acc[t] = zero;

  const int KT     = (K + 31) >> 5;
  const int KTfull = TAIL ? (KT - 1) : KT;

  for (int kt = 0; kt < KTfull; ++kt) {
    const int k0 = kt << 5;
    if (kt + 1 < KT) __builtin_prefetch(Arow + k0 + 32, 0, 3); // global_prefetch

    float4 f10 = *(const float4*)(Arow + k0);
    float4 f11 = *(const float4*)(Arow + k0 + 4);
    float4 f20 = *(const float4*)(Arow + k0 + 16);
    float4 f21 = *(const float4*)(Arow + k0 + 20);
    v16h a;
    cvt_a_frag(a, f10, f11, f20, f21);

    const _Float16* wpk = Wp + ((size_t)kt * NT_C) * 512 + lane * 16;
    v16h b[NT_C];
#pragma unroll
    for (int t = 0; t < NT_C; ++t) b[t] = *(const v16h*)(wpk + (size_t)t * 512);
#pragma unroll
    for (int t = 0; t < NT_C; ++t)
      acc[t] = __builtin_amdgcn_wmma_f32_16x16x32_f16(
          false, a, false, b[t], (short)0, acc[t], false, false);
  }

  if (TAIL) {  // final half tile: only first 16 of 32 K values valid
    const int k0 = KTfull << 5;
    float4 fz = make_float4(0.f, 0.f, 0.f, 0.f);
    float4 f10 = *(const float4*)(Arow + k0);
    float4 f11 = *(const float4*)(Arow + k0 + 4);
    v16h a;
    cvt_a_frag(a, f10, f11, fz, fz);
    const _Float16* wpk = Wp + ((size_t)KTfull * NT_C) * 512 + lane * 16;
    v16h b[NT_C];
#pragma unroll
    for (int t = 0; t < NT_C; ++t) b[t] = *(const v16h*)(wpk + (size_t)t * 512);
#pragma unroll
    for (int t = 0; t < NT_C; ++t)
      acc[t] = __builtin_amdgcn_wmma_f32_16x16x32_f16(
          false, a, false, b[t], (short)0, acc[t], false, false);
  }

  // epilogue: lane -> col (t*16 + ncol), VGPR r -> row row0 + r + khalf*8
#pragma unroll
  for (int t = 0; t < NT_C; ++t) {
    const int n  = (t << 4) + ncol;
    const float bv = bias ? bias[n] : 0.0f;
#pragma unroll
    for (int r = 0; r < 8; ++r) {
      const int m = row0 + r + (khalf << 3);
      if (m < M) {
        float v = acc[t][r] + bv;
        if (act == 1) v = seluf(v);
        C[(size_t)m * N + n] = v;
      }
    }
  }
}

// ---------------- elementwise / reduction kernels ----------------
__global__ void fill_kernel(float* p, float v, long n) {
  long id = (long)blockIdx.x * blockDim.x + threadIdx.x;
  if (id < n) p[id] = v;
}

// coalesced linear sweep + LDS float atomics, one global atomic per column per block
__global__ void bn_stats_kernel(const float* __restrict__ h, float* __restrict__ stats,
                                long total, int C, long elemsPerBlock) {
  __shared__ float ls[80], lq[80];
  if ((int)threadIdx.x < C) { ls[threadIdx.x] = 0.f; lq[threadIdx.x] = 0.f; }
  __syncthreads();
  long start = (long)blockIdx.x * elemsPerBlock;
  long end   = start + elemsPerBlock; if (end > total) end = total;
  long i = start + threadIdx.x;
  int  c = (int)(i % C);
  const int step = (int)(blockDim.x % C);
  for (; i < end; i += blockDim.x) {
    float v = h[i];
    atomicAdd(&ls[c], v);
    atomicAdd(&lq[c], v * v);
    c += step; if (c >= C) c -= C;
  }
  __syncthreads();
  if ((int)threadIdx.x < C) {
    atomicAdd(&stats[threadIdx.x], ls[threadIdx.x]);
    atomicAdd(&stats[C + threadIdx.x], lq[threadIdx.x]);
  }
}

__global__ void bn_apply_kernel(float* h, const float* __restrict__ stats,
                                const float* __restrict__ g, const float* __restrict__ bb,
                                long M, int C) {
  long id = (long)blockIdx.x * blockDim.x + threadIdx.x;
  if (id >= M * C) return;
  int c = (int)(id % C);
  float mean = stats[c] / (float)M;
  float var  = stats[C + c] / (float)M - mean * mean;
  h[id] = (h[id] - mean) * rsqrtf(var + EPSBN) * g[c] + bb[c];
}

__global__ void gather_edge_cond_kernel(const float* __restrict__ x, const float* __restrict__ u,
                                        const int* __restrict__ src, const int* __restrict__ dst,
                                        const int* __restrict__ batch, float* __restrict__ cond,
                                        long E) {
  long id = (long)blockIdx.x * blockDim.x + threadIdx.x;
  if (id >= E * 192) return;
  long e = id / 192; int j = (int)(id % 192);
  float v;
  if (j < 64)       v = x[(size_t)src[e] * 64 + j];
  else if (j < 128) v = x[(size_t)dst[e] * 64 + (j - 64)];
  else              v = u[(size_t)batch[src[e]] * 64 + (j - 128)];
  cond[id] = v;
}

__global__ void film_kernel(const float* __restrict__ ga, const float* __restrict__ be,
                            float* ea, long n) {
  long id = (long)blockIdx.x * blockDim.x + threadIdx.x;
  if (id < n) ea[id] = ga[id] * ea[id] + be[id];
}

__global__ void deg_kernel(const int* __restrict__ dst, float* deg, long E) {
  long id = (long)blockIdx.x * blockDim.x + threadIdx.x;
  if (id < E) atomicAdd(&deg[dst[id]], 1.f);
}

__global__ void loop_sum_kernel(const float* __restrict__ ea, const int* __restrict__ dst,
                                float* loopa, long E) {
  long id = (long)blockIdx.x * blockDim.x + threadIdx.x;
  if (id >= E * 32) return;
  long e = id / 32; int c = (int)(id % 32);
  atomicAdd(&loopa[(size_t)dst[e] * 32 + c], ea[id]);
}

__global__ void loop_div_kernel(float* loopa, const float* __restrict__ deg, long n32) {
  long id = (long)blockIdx.x * blockDim.x + threadIdx.x;
  if (id >= n32) return;
  float d = deg[id / 32];
  loopa[id] /= (d > 1.f ? d : 1.f);
}

__global__ void gat_logit_kernel(const float* __restrict__ xl, const float* __restrict__ xr,
                                 const float* __restrict__ ea2, const int* __restrict__ src,
                                 const int* __restrict__ dst, const float* __restrict__ att,
                                 float* __restrict__ logit, long E, long N) {
  long id = (long)blockIdx.x * blockDim.x + threadIdx.x;
  if (id >= (E + N) * HEADS) return;
  int h = (int)(id % HEADS); long i = id / HEADS;
  int s, d;
  if (i < E) { s = src[i]; d = dst[i]; } else { s = d = (int)(i - E); }
  const float* me = ea2 + (size_t)i * 80 + h * 16;
  const float* ml = xl  + (size_t)s * 80 + h * 16;
  const float* mr = xr  + (size_t)d * 80 + h * 16;
  float acc = 0.f;
#pragma unroll
  for (int c = 0; c < 16; ++c) {
    float m = ml[c] + mr[c] + me[c];
    float lr = m > 0.f ? m : 0.2f * m;
    acc += lr * att[h * 16 + c];
  }
  logit[id] = acc;
}

__global__ void seg_max_kernel(const float* __restrict__ logit, const int* __restrict__ dst,
                               float* mx, long E, long N) {
  long id = (long)blockIdx.x * blockDim.x + threadIdx.x;
  if (id >= (E + N) * HEADS) return;
  int h = (int)(id % HEADS); long i = id / HEADS;
  int d = (i < E) ? dst[i] : (int)(i - E);
  atomicMaxF(&mx[(size_t)d * HEADS + h], logit[id]);
}

__global__ void exp_den_kernel(const float* __restrict__ logit, const int* __restrict__ dst,
                               const float* __restrict__ mx, float* __restrict__ ex,
                               float* den, long E, long N) {
  long id = (long)blockIdx.x * blockDim.x + threadIdx.x;
  if (id >= (E + N) * HEADS) return;
  int h = (int)(id % HEADS); long i = id / HEADS;
  int d = (i < E) ? dst[i] : (int)(i - E);
  float e = expf(logit[id] - mx[(size_t)d * HEADS + h]);
  ex[id] = e;
  atomicAdd(&den[(size_t)d * HEADS + h], e);
}

__global__ void gat_out_kernel(const float* __restrict__ xl, const int* __restrict__ src,
                               const int* __restrict__ dst, const float* __restrict__ ex,
                               const float* __restrict__ den, float* out, long E, long N) {
  long id = (long)blockIdx.x * blockDim.x + threadIdx.x;
  if (id >= (E + N) * 80) return;
  long i = id / 80; int c = (int)(id % 80); int h = c >> 4;
  int s, d;
  if (i < E) { s = src[i]; d = dst[i]; } else { s = d = (int)(i - E); }
  float alpha = ex[i * HEADS + h] / den[(size_t)d * HEADS + h];
  atomicAdd(&out[(size_t)d * 80 + c], alpha * xl[(size_t)s * 80 + c]);
}

__global__ void bias_act_kernel(float* out, const float* __restrict__ bias, long n, int relu) {
  long id = (long)blockIdx.x * blockDim.x + threadIdx.x;
  if (id >= n) return;
  float v = out[id] + bias[id % 80];
  if (relu) v = v > 0.f ? v : 0.f;
  out[id] = v;
}

__global__ void gather_node_cond_kernel(const float* __restrict__ h, const float* __restrict__ x,
                                        const float* __restrict__ u, const int* __restrict__ batch,
                                        float* cond, long N) {
  long id = (long)blockIdx.x * blockDim.x + threadIdx.x;
  if (id >= N * 208) return;
  long n = id / 208; int j = (int)(id % 208);
  float v;
  if (j < 80)       v = h[n * 80 + j];
  else if (j < 144) v = x[n * 64 + (j - 80)];
  else              v = u[(size_t)batch[n] * 64 + (j - 144)];
  cond[id] = v;
}

__global__ void pna_stats_kernel(const float* __restrict__ x, const int* __restrict__ batch,
                                 float* meanS, float* sqS, float* mxS, float* mnS, float* cnt,
                                 long N) {
  long id = (long)blockIdx.x * blockDim.x + threadIdx.x;
  if (id >= N * 64) return;
  long n = id / 64; int c = (int)(id % 64);
  int b = batch[n];
  float v = x[id];
  atomicAdd(&meanS[(size_t)b * 64 + c], v);
  atomicAdd(&sqS[(size_t)b * 64 + c], v * v);
  atomicMaxF(&mxS[(size_t)b * 64 + c], v);
  atomicMinF(&mnS[(size_t)b * 64 + c], v);
  if (c == 0) atomicAdd(&cnt[b], 1.f);
}

__global__ void gather_glob_kernel(const float* __restrict__ u, const float* __restrict__ meanS,
                                   const float* __restrict__ sqS, const float* __restrict__ mxS,
                                   const float* __restrict__ mnS, const float* __restrict__ cnt,
                                   float* cond, long B) {
  long id = (long)blockIdx.x * blockDim.x + threadIdx.x;
  if (id >= B * 320) return;
  long b = id / 320; int j = (int)(id % 320);
  float v;
  if (j < 64) v = u[b * 64 + j];
  else {
    int c = (j - 64) % 64, reg = (j - 64) / 64;
    float cn = cnt[b] > 1.f ? cnt[b] : 1.f;
    float me = meanS[b * 64 + c] / cn;
    if (reg == 0) v = me;
    else if (reg == 1) {
      float d = sqS[b * 64 + c] / cn - me * me;
      v = sqrtf((d > 0.f ? d : 0.f) + EPSBN);
    } else if (reg == 2) v = mxS[b * 64 + c];
    else v = mnS[b * 64 + c];
  }
  cond[id] = v;
}

// ---------------- host-side orchestration ----------------
struct MlpP { const float *W0,*W1,*W2,*bb0,*bb1,*b0,*b1,*b2,*g0,*g1; int K,dout; };
struct GatP { const float *We,*Wl,*Wr,*att,*bias,*bl,*br; };

static inline int nblk(long n, int t) { return (int)((n + t - 1) / t); }

static void gemm(const float* A, const float* W, const float* b, float* C,
                 int M, int K, int N, int act, _Float16* wpack, hipStream_t s) {
  const int KT = (K + 31) / 32, NT = (N + 15) / 16;
  const long total = (long)KT * NT * 512;
  pack_w_kernel<<<nblk(total, 256), 256, 0, s>>>(W, wpack, K, N, NT, total);
  const int grid = nblk(M, 128);
  const bool tail = (K & 31) != 0;
  if (NT == 5) {
    if (tail) gemm_wmma_kernel<5, true ><<<grid, 256, 0, s>>>(A, wpack, b, C, M, K, act);
    else      gemm_wmma_kernel<5, false><<<grid, 256, 0, s>>>(A, wpack, b, C, M, K, act);
  } else if (NT == 4) {
    if (tail) gemm_wmma_kernel<4, true ><<<grid, 256, 0, s>>>(A, wpack, b, C, M, K, act);
    else      gemm_wmma_kernel<4, false><<<grid, 256, 0, s>>>(A, wpack, b, C, M, K, act);
  } else {
    if (tail) gemm_wmma_kernel<2, true ><<<grid, 256, 0, s>>>(A, wpack, b, C, M, K, act);
    else      gemm_wmma_kernel<2, false><<<grid, 256, 0, s>>>(A, wpack, b, C, M, K, act);
  }
}

static void batchnorm(float* h, const float* g, const float* bb, float* stats,
                      long M, int C, hipStream_t s) {
  fill_kernel<<<1, 256, 0, s>>>(stats, 0.f, 2 * C);
  const long total = M * C;
  const int blocks = 1024;
  const long epb = (total + blocks - 1) / blocks;
  bn_stats_kernel<<<blocks, 256, 0, s>>>(h, stats, total, C, epb);
  bn_apply_kernel<<<nblk(total, 256), 256, 0, s>>>(h, stats, g, bb, M, C);
}

static void run_mlp(const MlpP& p, const float* Ain, int M, float* out,
                    float* t1, float* t2, float* stats, _Float16* wpack, hipStream_t s) {
  gemm(Ain, p.W0, p.b0, t1, M, p.K, Hh, 1, wpack, s);
  batchnorm(t1, p.g0, p.bb0, stats, M, Hh, s);
  gemm(t1, p.W1, p.b1, t2, M, Hh, Hh, 1, wpack, s);
  batchnorm(t2, p.g1, p.bb1, stats, M, Hh, s);
  gemm(t2, p.W2, p.b2, out, M, Hh, p.dout, 0, wpack, s);
}

static void run_gat(const GatP& g, const float* xin, int DIN,
                    const int* src, const int* dst,
                    const float* ea, const float* loopa,
                    float* xl, float* xr, float* ea2, float* logit, float* ex,
                    float* mx, float* den, float* out, int relu,
                    _Float16* wpack, hipStream_t s) {
  gemm(xin, g.Wl, g.bl, xl, Nn, DIN, Hh, 0, wpack, s);
  gemm(xin, g.Wr, g.br, xr, Nn, DIN, Hh, 0, wpack, s);
  gemm(ea, g.We, nullptr, ea2, Ee, DEe, Hh, 0, wpack, s);
  gemm(loopa, g.We, nullptr, ea2 + (size_t)Ee * Hh, Nn, DEe, Hh, 0, wpack, s);
  long tot = (long)E2 * HEADS;
  gat_logit_kernel<<<nblk(tot, 256), 256, 0, s>>>(xl, xr, ea2, src, dst, g.att, logit, Ee, Nn);
  fill_kernel<<<nblk((long)Nn * HEADS, 256), 256, 0, s>>>(mx, -3.0e38f, (long)Nn * HEADS);
  seg_max_kernel<<<nblk(tot, 256), 256, 0, s>>>(logit, dst, mx, Ee, Nn);
  fill_kernel<<<nblk((long)Nn * HEADS, 256), 256, 0, s>>>(den, 0.f, (long)Nn * HEADS);
  exp_den_kernel<<<nblk(tot, 256), 256, 0, s>>>(logit, dst, mx, ex, den, Ee, Nn);
  fill_kernel<<<nblk((long)Nn * 80, 256), 256, 0, s>>>(out, 0.f, (long)Nn * 80);
  gat_out_kernel<<<nblk((long)E2 * 80, 256), 256, 0, s>>>(xl, src, dst, ex, den, out, Ee, Nn);
  bias_act_kernel<<<nblk((long)Nn * 80, 256), 256, 0, s>>>(out, g.bias, (long)Nn * 80, relu);
}

extern "C" void kernel_launch(void* const* d_in, const int* in_sizes, int n_in,
                              void* d_out, int out_size, void* d_ws, size_t ws_size,
                              hipStream_t stream) {
  (void)in_sizes; (void)n_in; (void)out_size; (void)ws_size;
  const float* x_in  = (const float*)d_in[0];
  const int*   ei    = (const int*)d_in[1];
  const float* ea_in = (const float*)d_in[2];
  const float* u_in  = (const float*)d_in[3];
  const int*   batch = (const int*)d_in[4];
  const int* src = ei;
  const int* dst = ei + Ee;

  // ---- workspace bump allocator (256B aligned) ----
  char* wp = (char*)d_ws;
  auto alloc = [&](size_t nf) -> float* {
    float* p = (float*)wp;
    wp += ((nf * sizeof(float) + 255) & ~(size_t)255);
    return p;
  };
  float* xcur  = alloc((size_t)Nn * 64);
  float* eacur = alloc((size_t)Ee * 32);
  float* ucur  = alloc((size_t)Bb * 64);
  float* condE = alloc((size_t)Ee * 192);   // reused for node cond (N*208 < E*192)
  float* tmp1  = alloc((size_t)E2 * 80);    // mlp h1 / gat ea2
  float* tmp2  = alloc((size_t)Ee * 80);    // mlp h2
  float* gout  = alloc((size_t)Ee * 32);
  float* bout  = alloc((size_t)Ee * 32);
  float* xl    = alloc((size_t)Nn * 80);
  float* xr    = alloc((size_t)Nn * 80);
  float* hbuf1 = alloc((size_t)Nn * 80);
  float* hbuf2 = alloc((size_t)Nn * 80);
  float* loopa = alloc((size_t)Nn * 32);
  float* deg   = alloc((size_t)Nn);
  float* logit = alloc((size_t)E2 * HEADS);
  float* exb   = alloc((size_t)E2 * HEADS);
  float* mxb   = alloc((size_t)Nn * HEADS);
  float* denb  = alloc((size_t)Nn * HEADS);
  float* stats = alloc(2 * Hh);
  float* meanS = alloc((size_t)Bb * 64);
  float* sqS   = alloc((size_t)Bb * 64);
  float* mxS   = alloc((size_t)Bb * 64);
  float* mnS   = alloc((size_t)Bb * 64);
  float* cnt   = alloc((size_t)Bb);
  float* gcond = alloc((size_t)Bb * 320);
  _Float16* wpack = (_Float16*)alloc(16384);  // max KT(10)*NT(5)*512 halfs

  hipMemcpyAsync(xcur, x_in, (size_t)Nn * 64 * sizeof(float), hipMemcpyDeviceToDevice, stream);
  hipMemcpyAsync(eacur, ea_in, (size_t)Ee * 32 * sizeof(float), hipMemcpyDeviceToDevice, stream);
  hipMemcpyAsync(ucur, u_in, (size_t)Bb * 64 * sizeof(float), hipMemcpyDeviceToDevice, stream);

  auto F = [&](int i) -> const float* { return (const float*)d_in[i]; };
  auto mkMlp = [&](int base, int K, int dout) {
    MlpP m;
    m.W0 = F(base + 0); m.W1 = F(base + 1); m.W2 = F(base + 2);
    m.bb0 = F(base + 3); m.bb1 = F(base + 4);
    m.b0 = F(base + 5); m.b1 = F(base + 6); m.b2 = F(base + 7);
    m.g0 = F(base + 8); m.g1 = F(base + 9);
    m.K = K; m.dout = dout; return m;
  };
  auto mkGat = [&](int base) {
    GatP g;
    g.We = F(base + 0); g.Wl = F(base + 1); g.Wr = F(base + 2);
    g.att = F(base + 3); g.bias = F(base + 4); g.bl = F(base + 5); g.br = F(base + 6);
    return g;
  };

  for (int d = 0; d < DEPTH; ++d) {
    const int base = 5 + d * 54; // sorted pytree: edge.beta, edge.gamma, glob.mlp, node.gat1, node.gat2, node.mlp
    MlpP mBeta  = mkMlp(base + 0, 192, DEe);
    MlpP mGamma = mkMlp(base + 10, 192, DEe);
    MlpP mGlob  = mkMlp(base + 20, 320, DGg);
    GatP gat1   = mkGat(base + 30);
    GatP gat2   = mkGat(base + 37);
    MlpP mNode  = mkMlp(base + 44, 208, Dd);

    // ---- edge FILM ----
    gather_edge_cond_kernel<<<nblk((long)Ee * 192, 256), 256, 0, stream>>>(
        xcur, ucur, src, dst, batch, condE, Ee);
    run_mlp(mGamma, condE, Ee, gout, tmp1, tmp2, stats, wpack, stream);
    run_mlp(mBeta,  condE, Ee, bout, tmp1, tmp2, stats, wpack, stream);
    film_kernel<<<nblk((long)Ee * 32, 256), 256, 0, stream>>>(gout, bout, eacur, (long)Ee * 32);

    // ---- self-loop edge_attr (segment mean over dst) ----
    fill_kernel<<<nblk((long)Nn, 256), 256, 0, stream>>>(deg, 0.f, (long)Nn);
    fill_kernel<<<nblk((long)Nn * 32, 256), 256, 0, stream>>>(loopa, 0.f, (long)Nn * 32);
    deg_kernel<<<nblk((long)Ee, 256), 256, 0, stream>>>(dst, deg, Ee);
    loop_sum_kernel<<<nblk((long)Ee * 32, 256), 256, 0, stream>>>(eacur, dst, loopa, Ee);
    loop_div_kernel<<<nblk((long)Nn * 32, 256), 256, 0, stream>>>(loopa, deg, (long)Nn * 32);

    // ---- node block: GATv2 x2 + MLP ----
    run_gat(gat1, xcur, Dd, src, dst, eacur, loopa,
            xl, xr, tmp1, logit, exb, mxb, denb, hbuf1, 1, wpack, stream);
    run_gat(gat2, hbuf1, Hh, src, dst, eacur, loopa,
            xl, xr, tmp1, logit, exb, mxb, denb, hbuf2, 0, wpack, stream);
    gather_node_cond_kernel<<<nblk((long)Nn * 208, 256), 256, 0, stream>>>(
        hbuf2, xcur, ucur, batch, condE, Nn);
    run_mlp(mNode, condE, Nn, xcur, tmp1, tmp2, stats, wpack, stream);

    // ---- global PNA ----
    fill_kernel<<<nblk((long)Bb * 64, 256), 256, 0, stream>>>(meanS, 0.f, (long)Bb * 64);
    fill_kernel<<<nblk((long)Bb * 64, 256), 256, 0, stream>>>(sqS, 0.f, (long)Bb * 64);
    fill_kernel<<<nblk((long)Bb * 64, 256), 256, 0, stream>>>(mxS, -3.0e38f, (long)Bb * 64);
    fill_kernel<<<nblk((long)Bb * 64, 256), 256, 0, stream>>>(mnS, 3.0e38f, (long)Bb * 64);
    fill_kernel<<<1, 256, 0, stream>>>(cnt, 0.f, (long)Bb);
    pna_stats_kernel<<<nblk((long)Nn * 64, 256), 256, 0, stream>>>(
        xcur, batch, meanS, sqS, mxS, mnS, cnt, Nn);
    gather_glob_kernel<<<nblk((long)Bb * 320, 256), 256, 0, stream>>>(
        ucur, meanS, sqS, mxS, mnS, cnt, gcond, Bb);
    run_mlp(mGlob, gcond, Bb, ucur, tmp1, tmp2, stats, wpack, stream);
  }

  hipMemcpyAsync(d_out, ucur, (size_t)Bb * 64 * sizeof(float), hipMemcpyDeviceToDevice, stream);
}
